// NN_51780125721047
// MI455X (gfx1250) — matrix operand order repaired and checked by reading, not compiled, not run on hardware
//
#include <hip/hip_runtime.h>
#include <hip/hip_bf16.h>

// ---------------- problem constants ----------------
#define QN 8192      // queries (rows of x)
#define TN 32768     // train points
#define DD 128       // feature dim
#define NSPLIT 16    // N-chunks across blockIdx.y
#define NTILE 64     // train points staged per TDM tile
#define LROW 136     // padded LDS row stride in elements (128 + 8) -> 272B

typedef __attribute__((ext_vector_type(16))) __bf16 v16bf;
typedef __attribute__((ext_vector_type(8)))  __bf16 v8bf;
typedef __attribute__((ext_vector_type(8)))  float  v8f;
typedef __attribute__((ext_vector_type(4)))  unsigned int u32x4;
typedef __attribute__((ext_vector_type(8)))  int i32x8;
typedef __attribute__((ext_vector_type(4)))  int i32x4;

union BFrag { v16bf v; v8bf h[2]; };

// A-matrix 16x32 bf16 per-lane layout (ISA 7.12.2):
//   lane L (half = L>>4, m = L&15): elems 0..7  = K [half*8, +8), elems 8..15 = K [16+half*8, +8)
__device__ inline v16bf load_fragA(const __bf16* base, int row, int kc, int half) {
    const int k0 = kc * 32 + half * 8;
    const v8bf* p = (const v8bf*)(base + (size_t)row * DD + k0);
    BFrag f;
    f.h[0] = p[0];   // K = k0 .. k0+7
    f.h[1] = p[2];   // K = k0+16 .. k0+23
    return f.v;
}

// B-matrix 32x16 bf16 per-lane layout: lane (half, n): elems 0..7 = K [half*8, +8), elems 8..15 = K [16+half*8, +8)
// Read from the padded LDS tile (row stride LROW elements).
__device__ inline v16bf load_fragB_lds(const unsigned short* base, int row, int koff) {
    const v8bf* p = (const v8bf*)(base + row * LROW + koff);
    BFrag f;
    f.h[0] = p[0];
    f.h[1] = p[2];
    return f.v;
}

// ---- Tensor Data Mover: stage a [NTILE rows x 128 K] bf16 tile into LDS with row padding ----
// D# packed per CDNA5 ISA ch.8: data_size=2B, pad after every 64 DWORDs (256B row) by 4 DWORDs (16B)
// so the LDS row stride becomes 272B = LROW elements.
// This toolchain (clang-23 / therock-10.0) exposes the 6-arg builtin:
//   (uint32x4 g0, int32x8 g1, int32x4 g2, int32x4 g3, int32x8 extra, i32 cpol)
__device__ inline void tdm_load_tile(const unsigned short* gsrc, unsigned lds_addr) {
    unsigned long long ga = (unsigned long long)(size_t)gsrc;
    u32x4 g0;
    g0[0] = 1u;                                         // count=1 (valid user descriptor)
    g0[1] = lds_addr;                                   // LDS byte address
    g0[2] = (unsigned)(ga & 0xFFFFFFFFu);               // global_addr[31:0]
    g0[3] = ((unsigned)(ga >> 32) & 0x01FFFFFFu)        // global_addr[56:32]
            | (2u << 30);                               // type=2 ("image")
    i32x8 g1;
    g1[0] = (int)((1u << 16)      // data_size = 1 -> 2 bytes
                | (1u << 20)      // pad_enable
                | (5u << 22)      // pad_interval: 64 DWORDs (=256B = one row)
                | (3u << 25));    // pad_amount: 4 DWORDs (=16B)
    g1[1] = (int)(128u << 16);    // tensor_dim0 = 128 (bits 79:48, low16 here)
    g1[2] = 0;                    // tensor_dim0 hi16 = 0, tensor_dim1 low16 = 0
    g1[3] = (int)(16u | (128u << 16)); // tensor_dim1 hi16 (=1<<20 rows), tile_dim0 = 128
    g1[4] = (int)NTILE;           // tile_dim1 = 64, tile_dim2 = 0 (unused)
    g1[5] = 128;                  // tensor_dim0_stride = 128 (low 32)
    g1[6] = 0;                    // stride hi, tensor_dim1_stride low16 (unused for 2D)
    g1[7] = 0;
    i32x4 g2 = {0, 0, 0, 0};      // dims 2/3 unused (tile_dim3 = 0)
    i32x4 g3 = {0, 0, 0, 0};      // dims 3/4 unused
    i32x8 g4 = {0, 0, 0, 0, 0, 0, 0, 0};  // extra group (zero-fill, probe-verified)
    __builtin_amdgcn_tensor_load_to_lds(g0, g1, g2, g3, g4, 0);
}

// ---------------- kernel 1: fp32 -> bf16 hi/lo split ----------------
__global__ void split_bf16_kernel(const float* __restrict__ in,
                                  unsigned short* __restrict__ hi_u,
                                  unsigned short* __restrict__ lo_u,
                                  int n) {
    int i = blockIdx.x * blockDim.x + threadIdx.x;
    if (i >= n) return;
    __bf16* hi = (__bf16*)hi_u;
    __bf16* lo = (__bf16*)lo_u;
    float a = in[i];
    __bf16 h = (__bf16)a;
    hi[i] = h;
    lo[i] = (__bf16)(a - (float)h);
}

// ---------------- kernel 2: y2[j] = ||train_pts[j]||^2 (exact fp32) ----------------
__global__ void rownorm_kernel(const float* __restrict__ y, float* __restrict__ y2) {
    int r = blockIdx.x * blockDim.x + threadIdx.x;
    if (r >= TN) return;
    const float4* p = (const float4*)(y + (size_t)r * DD);
    float s = 0.f;
#pragma unroll 8
    for (int k = 0; k < DD / 4; ++k) {
        float4 v = p[k];
        s += v.x * v.x + v.y * v.y + v.z * v.z + v.w * v.w;
    }
    y2[r] = s;
}

// ---------------- kernel 3: TDM-staged WMMA GEMM + per-row argmin (partial) ----------------
__global__ __launch_bounds__(256) void nn_wmma_kernel(
    const unsigned short* __restrict__ xh_u, const unsigned short* __restrict__ xl_u,
    const unsigned short* __restrict__ yh_u, const unsigned short* __restrict__ yl_u,
    const float* __restrict__ y2,
    float* __restrict__ pval, int* __restrict__ pidx)
{
    // [buf][hi/lo][NTILE * LROW] padded bf16 tiles, double buffered: 68KB total
    __shared__ __align__(16) unsigned short smem[2][2][NTILE * LROW];

    const __bf16* xh = (const __bf16*)xh_u;
    const __bf16* xl = (const __bf16*)xl_u;

    const int lane = threadIdx.x & 31;
    const int wave = threadIdx.x >> 5;
    const int half = lane >> 4;
    const int l16  = lane & 15;
    const int m0   = blockIdx.x * 128 + wave * 16;   // 8 waves x 16 rows
    const int arow = m0 + l16;

    // Preload A fragments (hi/lo for 4 K-chunks) — resident for the whole N loop.
    v16bf ah[4], al[4];
#pragma unroll
    for (int kc = 0; kc < 4; ++kc) {
        ah[kc] = load_fragA(xh, arow, kc, half);
        al[kc] = load_fragA(xl, arow, kc, half);
    }

    float bestv[8];
    int   besti[8];
#pragma unroll
    for (int r = 0; r < 8; ++r) { bestv[r] = __builtin_inff(); besti[r] = 0; }

    const int nchunk = TN / NSPLIT;           // 2048
    const int nstart = blockIdx.y * nchunk;
    const int T = nchunk / NTILE;             // 32 tiles

    // Prologue: wave 0 stages tile 0 into buffer 0 via TDM (EXEC-independent, TENSORcnt-tracked).
    if (wave == 0) {
        tdm_load_tile(yh_u + (size_t)nstart * DD, (unsigned)(size_t)&smem[0][0][0]);
        tdm_load_tile(yl_u + (size_t)nstart * DD, (unsigned)(size_t)&smem[0][1][0]);
    }

    for (int t = 0; t < T; ++t) {
        const int buf = t & 1;
        const int n0 = nstart + t * NTILE;

        if (wave == 0) {
            if (t + 1 < T) {
                // Prefetch next tile into the other buffer, then wait only for the
                // current tile's 2 loads (the 2 just-issued stay in flight).
                tdm_load_tile(yh_u + (size_t)(n0 + NTILE) * DD,
                              (unsigned)(size_t)&smem[buf ^ 1][0][0]);
                tdm_load_tile(yl_u + (size_t)(n0 + NTILE) * DD,
                              (unsigned)(size_t)&smem[buf ^ 1][1][0]);
                __builtin_amdgcn_s_wait_tensorcnt(2);
            } else {
                __builtin_amdgcn_s_wait_tensorcnt(0);
            }
        }
        __syncthreads();   // tile t visible to all 8 waves

        const unsigned short* sh = smem[buf][0];
        const unsigned short* sl = smem[buf][1];

        // 4 subtiles of 16 train points; process 2 at a time -> independent WMMA chains.
#pragma unroll
        for (int s = 0; s < 4; s += 2) {
            const int r0 = s * 16 + l16;
            const int r1 = r0 + 16;
            v8f acc0 = {0.f, 0.f, 0.f, 0.f, 0.f, 0.f, 0.f, 0.f};
            v8f acc1 = {0.f, 0.f, 0.f, 0.f, 0.f, 0.f, 0.f, 0.f};
#pragma unroll
            for (int kc = 0; kc < 4; ++kc) {
                const int koff = kc * 32 + half * 8;
                v16bf bh0 = load_fragB_lds(sh, r0, koff);
                v16bf bl0 = load_fragB_lds(sl, r0, koff);
                v16bf bh1 = load_fragB_lds(sh, r1, koff);
                v16bf bl1 = load_fragB_lds(sl, r1, koff);
                acc0 = __builtin_amdgcn_wmma_f32_16x16x32_bf16(false, ah[kc], false, bh0,
                                                               (short)0, acc0, false, false);
                acc1 = __builtin_amdgcn_wmma_f32_16x16x32_bf16(false, ah[kc], false, bh1,
                                                               (short)0, acc1, false, false);
                acc0 = __builtin_amdgcn_wmma_f32_16x16x32_bf16(false, ah[kc], false, bl0,
                                                               (short)0, acc0, false, false);
                acc1 = __builtin_amdgcn_wmma_f32_16x16x32_bf16(false, ah[kc], false, bl1,
                                                               (short)0, acc1, false, false);
                acc0 = __builtin_amdgcn_wmma_f32_16x16x32_bf16(false, al[kc], false, bh0,
                                                               (short)0, acc0, false, false);
                acc1 = __builtin_amdgcn_wmma_f32_16x16x32_bf16(false, al[kc], false, bh1,
                                                               (short)0, acc1, false, false);
            }
            const int col0 = n0 + s * 16 + l16;
            const int col1 = col0 + 16;
            const float y20 = y2[col0];
            const float y21 = y2[col1];
#pragma unroll
            for (int r = 0; r < 8; ++r) {
                float d0 = fmaf(-2.0f, acc0[r], y20);
                float d1 = fmaf(-2.0f, acc1[r], y21);
                if (d0 < bestv[r]) { bestv[r] = d0; besti[r] = col0; }
                if (d1 < bestv[r]) { bestv[r] = d1; besti[r] = col1; }
            }
        }
        __syncthreads();   // everyone done reading smem[buf] before it is re-staged
    }

    // Reduce across the 16 lanes of each half (masks < 16 keep wave32 halves separate).
#pragma unroll
    for (int r = 0; r < 8; ++r) {
#pragma unroll
        for (int m = 1; m < 16; m <<= 1) {
            float ov = __shfl_xor(bestv[r], m, 32);
            int   oi = __shfl_xor(besti[r], m, 32);
            if (ov < bestv[r] || (ov == bestv[r] && oi < besti[r])) {
                bestv[r] = ov; besti[r] = oi;
            }
        }
    }

    if (l16 == 0) {
        // lane 0 holds rows m0+0..7 (half=0); lane 16 holds rows m0+8..15 (half=1)
#pragma unroll
        for (int r = 0; r < 8; ++r) {
            int rowg = m0 + half * 8 + r;
            pval[rowg * NSPLIT + blockIdx.y] = bestv[r];
            pidx[rowg * NSPLIT + blockIdx.y] = besti[r];
        }
    }
}

// ---------------- kernel 4: reduce partials, gather label ----------------
__global__ void nn_reduce_kernel(const float* __restrict__ pval,
                                 const int* __restrict__ pidx,
                                 const float* __restrict__ labels,
                                 float* __restrict__ out) {
    int r = blockIdx.x * blockDim.x + threadIdx.x;
    if (r >= QN) return;
    float best = __builtin_inff();
    int   bi = 0;
    for (int c = 0; c < NSPLIT; ++c) {
        float v = pval[r * NSPLIT + c];
        if (v < best) { best = v; bi = pidx[r * NSPLIT + c]; }
    }
    out[r] = labels[bi];
}

// ---------------- host launcher ----------------
extern "C" void kernel_launch(void* const* d_in, const int* in_sizes, int n_in,
                              void* d_out, int out_size, void* d_ws, size_t ws_size,
                              hipStream_t stream) {
    const float* x  = (const float*)d_in[0];   // [QN, DD]
    const float* tp = (const float*)d_in[1];   // [TN, DD]
    const float* tl = (const float*)d_in[2];   // [TN]
    float* out = (float*)d_out;                // [QN]

    char* ws = (char*)d_ws;
    size_t off = 0;
    unsigned short* xh = (unsigned short*)(ws + off); off += (size_t)QN * DD * 2;
    unsigned short* xl = (unsigned short*)(ws + off); off += (size_t)QN * DD * 2;
    unsigned short* yh = (unsigned short*)(ws + off); off += (size_t)TN * DD * 2;
    unsigned short* yl = (unsigned short*)(ws + off); off += (size_t)TN * DD * 2;
    float* y2   = (float*)(ws + off);          off += (size_t)TN * 4;
    float* pval = (float*)(ws + off);          off += (size_t)QN * NSPLIT * 4;
    int*   pidx = (int*)(ws + off);            off += (size_t)QN * NSPLIT * 4;
    (void)off; (void)ws_size; (void)in_sizes; (void)n_in; (void)out_size;

    // 1) bf16 hi/lo splits
    {
        int n = QN * DD;
        split_bf16_kernel<<<(n + 255) / 256, 256, 0, stream>>>(x, xh, xl, n);
    }
    {
        int n = TN * DD;
        split_bf16_kernel<<<(n + 255) / 256, 256, 0, stream>>>(tp, yh, yl, n);
    }
    // 2) row norms of train points (exact fp32)
    rownorm_kernel<<<(TN + 255) / 256, 256, 0, stream>>>(tp, y2);

    // 3) TDM-staged WMMA distance + partial argmin
    dim3 grid(QN / 128, NSPLIT);
    nn_wmma_kernel<<<grid, 256, 0, stream>>>(xh, xl, yh, yl, y2, pval, pidx);

    // 4) final reduce + label gather
    nn_reduce_kernel<<<(QN + 255) / 256, 256, 0, stream>>>(pval, pidx, tl, out);
}
